// AttentionForQuantizer_70076686402093
// MI455X (gfx1250) — compile-verified, matrix operand order checked by reading, not run on hardware
//
#include <hip/hip_runtime.h>
#include <hip/hip_bf16.h>

// MI455X / gfx1250 implementation.
// - All GEMMs via v_wmma_f32_16x16x32_bf16 (f32 accumulate).
// - logits (the 537MB / 68.7 GFLOP op) fuses per-row argmax: written once,
//   never re-read. z_q == z_q_2 == v[argmax] in forward value (STE cancels).
// - k matrix staged per-block into LDS (double-buffered with gfx1250
//   GLOBAL_LOAD_ASYNC_TO_LDS_B128) so k is read from L2 once per block
//   instead of once per wave (4GB -> 512MB of L2 traffic).
// - LDS tiles padded to 264 bf16/row: 132 dwords == 4 (mod 64 banks), so a
//   16-lane ds_load_b128 fragment read tiles all 64 banks exactly (2-cycle
//   optimum) while preserving 16B alignment.

#define RMS_EPS 1e-5f

typedef __attribute__((ext_vector_type(16))) __bf16 v16bf;
typedef __attribute__((ext_vector_type(8)))  __bf16 bf16x8;
typedef __attribute__((ext_vector_type(8)))  float  v8f;
typedef __attribute__((ext_vector_type(4)))  float  v4f;
typedef __attribute__((ext_vector_type(4)))  int    v4i;

#if defined(__gfx1250__) && __has_builtin(__builtin_amdgcn_global_load_async_to_lds_b128) && __has_builtin(__builtin_amdgcn_s_wait_asynccnt)
#define HAVE_ASYNC_LDS 1
typedef __attribute__((address_space(1))) v4i* gv4i_p;   // global
typedef __attribute__((address_space(3))) v4i* lv4i_p;   // LDS
#else
#define HAVE_ASYNC_LDS 0
#endif

static __device__ __forceinline__ __bf16 f2bf(float f) {
  return (__bf16)f;   // fptrunc, RNE -> hardware cvt
}

// Fragment loader for 16x16x32 bf16 WMMA (A rows / B columns, symmetric
// layout): per K-step, lane (m = lane&15, half = lane>>4) holds two
// contiguous 8-element chunks at k0+8*half and k0+16+8*half.
static __device__ __forceinline__ v16bf load_frag_bf16(const __bf16* row, int k0, int half) {
  bf16x8 c0 = *(const bf16x8*)(row + k0 + 8 * half);
  bf16x8 c1 = *(const bf16x8*)(row + k0 + 16 + 8 * half);
  v16bf a;
#pragma unroll
  for (int i = 0; i < 8; ++i) { a[i] = c0[i]; a[8 + i] = c1[i]; }
  return a;
}

static __device__ __forceinline__ v8f wmma_bf16(v16bf a, v16bf b, v8f c) {
  return __builtin_amdgcn_wmma_f32_16x16x32_bf16(false, a, false, b, (short)0, c, false, false);
}

// Copy one 16B unit global->LDS (async on gfx1250 when available).
static __device__ __forceinline__ void copy16_g2l(const __bf16* g, __bf16* l) {
#if HAVE_ASYNC_LDS
  __builtin_amdgcn_global_load_async_to_lds_b128(
      (gv4i_p)(uintptr_t)g,
      (lv4i_p)(unsigned)(uintptr_t)l,
      0, 0);
#else
  *(v4i*)l = *(const v4i*)g;
#endif
}

static __device__ __forceinline__ void wait_async_lds() {
#if HAVE_ASYNC_LDS
  __builtin_amdgcn_s_wait_asynccnt(0);
#endif
}

// ---------------------------------------------------------------------------
// Kernel 1: weights f32 [K=256][N=256] -> bf16 transposed [N][K].
// Consecutive threads write consecutive k: coalesced 2B stores; strided reads
// are L2-cached (one-time 384KB).
__global__ void k_convert_w(const float* __restrict__ wq, const float* __restrict__ wk,
                            const float* __restrict__ wv,
                            __bf16* __restrict__ wTq, __bf16* __restrict__ wTk,
                            __bf16* __restrict__ wTv) {
  int gid = blockIdx.x * blockDim.x + threadIdx.x;   // 0 .. 3*65536-1
  int mat = gid >> 16;
  int e   = gid & 0xffff;
  int n   = e >> 8;    // out-dim
  int k   = e & 255;   // in-dim
  const float* w = (mat == 0) ? wq : ((mat == 1) ? wk : wv);
  __bf16* o      = (mat == 0) ? wTq : ((mat == 1) ? wTk : wTv);
  o[n * 256 + k] = f2bf(w[k * 256 + n]);
}

// ---------------------------------------------------------------------------
// Kernel 2: y = rmsnorm(src @ W + b) * g * scale, output bf16.
// 256 threads = 8 waves; 16 rows/block. A-tile converted to bf16 in LDS ONCE,
// then each wave reads fragments via ds_load_b128.
__global__ void k_proj_norm(const float* __restrict__ src, const __bf16* __restrict__ wT,
                            const float* __restrict__ bias, const float* __restrict__ g,
                            __bf16* __restrict__ dst, float scale) {
  __shared__ alignas(16) __bf16 abf[16 * 264];   // padded rows: bank-tiling
  __shared__ float lds[16 * 260];
  const int tid  = threadIdx.x;
  const int wave = tid >> 5;
  const int lane = tid & 31;
  const int half = lane >> 4;
  const int lm   = lane & 15;
  const int row0 = blockIdx.x * 16;

  // --- stage: convert 16x256 f32 tile -> bf16 LDS, once per block ---
  {
    const int r = tid >> 4, j = tid & 15;
    const float* sp = src + (size_t)(row0 + r) * 256 + j * 16;
    v4f f0 = *(const v4f*)(sp);
    v4f f1 = *(const v4f*)(sp + 4);
    v4f f2 = *(const v4f*)(sp + 8);
    v4f f3 = *(const v4f*)(sp + 12);
    bf16x8 h0, h1;
#pragma unroll
    for (int i = 0; i < 4; ++i) {
      h0[i] = f2bf(f0[i]); h0[4 + i] = f2bf(f1[i]);
      h1[i] = f2bf(f2[i]); h1[4 + i] = f2bf(f3[i]);
    }
    *(bf16x8*)&abf[r * 264 + j * 16]     = h0;
    *(bf16x8*)&abf[r * 264 + j * 16 + 8] = h1;
  }
  __syncthreads();

  v16bf afrag[8];
#pragma unroll
  for (int ks = 0; ks < 8; ++ks) afrag[ks] = load_frag_bf16(&abf[lm * 264], ks * 32, half);

#pragma unroll
  for (int t = 0; t < 2; ++t) {
    const int n0 = (wave * 2 + t) * 16;
    const __bf16* brow = wT + (n0 + lm) * 256;
    v8f acc = {};
#pragma unroll
    for (int ks = 0; ks < 8; ++ks) {
      v16bf b = load_frag_bf16(brow, ks * 32, half);
      acc = wmma_bf16(afrag[ks], b, acc);
    }
    float bv = bias[n0 + lm];
#pragma unroll
    for (int r = 0; r < 8; ++r)
      lds[(r + 8 * half) * 260 + n0 + lm] = acc[r] + bv;
  }
  __syncthreads();

  // --- RMSNorm: 16 threads per row, 16 elements each ---
  const int row = tid >> 4;
  const int j   = tid & 15;
  float x[16];
  float s = 0.f;
#pragma unroll
  for (int i = 0; i < 16; ++i) { x[i] = lds[row * 260 + j * 16 + i]; s += x[i] * x[i]; }
#pragma unroll
  for (int off = 1; off < 16; off <<= 1) s += __shfl_xor(s, off, 32);
  float inv = rsqrtf(s * (1.0f / 256.0f) + RMS_EPS) * scale;
  __bf16* drow = dst + (size_t)(row0 + row) * 256;
#pragma unroll
  for (int i = 0; i < 16; ++i) {
    int c = j * 16 + i;
    drow[c] = f2bf(x[i] * inv * g[c]);
  }
}

// ---------------------------------------------------------------------------
// Kernel 3: v = src @ W + b, output f32 (no norm). Same LDS A-staging.
__global__ void k_proj_plain(const float* __restrict__ src, const __bf16* __restrict__ wT,
                             const float* __restrict__ bias, float* __restrict__ dst) {
  __shared__ alignas(16) __bf16 abf[16 * 264];
  const int tid  = threadIdx.x;
  const int wave = tid >> 5;
  const int lane = tid & 31;
  const int half = lane >> 4;
  const int lm   = lane & 15;
  const int row0 = blockIdx.x * 16;

  {
    const int r = tid >> 4, j = tid & 15;
    const float* sp = src + (size_t)(row0 + r) * 256 + j * 16;
    v4f f0 = *(const v4f*)(sp);
    v4f f1 = *(const v4f*)(sp + 4);
    v4f f2 = *(const v4f*)(sp + 8);
    v4f f3 = *(const v4f*)(sp + 12);
    bf16x8 h0, h1;
#pragma unroll
    for (int i = 0; i < 4; ++i) {
      h0[i] = f2bf(f0[i]); h0[4 + i] = f2bf(f1[i]);
      h1[i] = f2bf(f2[i]); h1[4 + i] = f2bf(f3[i]);
    }
    *(bf16x8*)&abf[r * 264 + j * 16]     = h0;
    *(bf16x8*)&abf[r * 264 + j * 16 + 8] = h1;
  }
  __syncthreads();

  v16bf afrag[8];
#pragma unroll
  for (int ks = 0; ks < 8; ++ks) afrag[ks] = load_frag_bf16(&abf[lm * 264], ks * 32, half);

#pragma unroll
  for (int t = 0; t < 2; ++t) {
    const int n0 = (wave * 2 + t) * 16;
    const __bf16* brow = wT + (n0 + lm) * 256;
    v8f acc = {};
#pragma unroll
    for (int ks = 0; ks < 8; ++ks) {
      v16bf b = load_frag_bf16(brow, ks * 32, half);
      acc = wmma_bf16(afrag[ks], b, acc);
    }
    float bv = bias[n0 + lm];
#pragma unroll
    for (int r = 0; r < 8; ++r)
      dst[(size_t)(row0 + r + 8 * half) * 256 + n0 + lm] = acc[r] + bv;
  }
}

// ---------------------------------------------------------------------------
// Kernel 4: logits = q_bf16 @ k_bf16^T (scale prefolded), fused argmax.
// Block = 8 waves x 16 rows = 128 rows. k staged through LDS in 64-row chunks,
// double-buffered with async loads-to-LDS; each block reads k exactly once.
#define KCH 64                 // k rows per chunk
#define KSTRIDE 264            // padded LDS row stride (bf16 elems)

__global__ void k_logits(const __bf16* __restrict__ qb, const __bf16* __restrict__ kb,
                         float* __restrict__ logits, int* __restrict__ amax) {
  __shared__ alignas(16) __bf16 kt[2][KCH * KSTRIDE];   // 2 x 33KB
  const int tid  = threadIdx.x;
  const int wave = tid >> 5;
  const int lane = tid & 31;
  const int half = lane >> 4;
  const int lm   = lane & 15;
  const int m0   = (blockIdx.x * 8 + wave) * 16;

  // q fragments resident in VGPRs for the whole kernel
  v16bf qf[8];
  const __bf16* qrow = qb + (size_t)(m0 + lm) * 256;
#pragma unroll
  for (int ks = 0; ks < 8; ++ks) qf[ks] = load_frag_bf16(qrow, ks * 32, half);

  float best[8];
  int   bidx[8];
#pragma unroll
  for (int r = 0; r < 8; ++r) { best[r] = -3.4e38f; bidx[r] = 0; }

  // stage chunk c into buffer buf: 64 rows x 512B = 32KB, 8 units/thread
  auto stage = [&](int c, int buf) {
    const __bf16* gsrc = kb + (size_t)c * KCH * 256;
#pragma unroll
    for (int i = 0; i < 8; ++i) {
      int u = tid + 256 * i;        // 16B unit, 0..2047
      int r = u >> 5;               // row in chunk
      int j = u & 31;               // 16B unit in row
      copy16_g2l(gsrc + r * 256 + j * 8, &kt[buf][r * KSTRIDE + j * 8]);
    }
  };

  stage(0, 0);
  wait_async_lds();
  __syncthreads();

  for (int c = 0; c < 4096 / KCH; ++c) {
    const int buf = c & 1;
    if (c + 1 < 4096 / KCH) stage(c + 1, buf ^ 1);   // overlap with compute

#pragma unroll
    for (int t = 0; t < KCH / 16; ++t) {
      const int n0 = c * KCH + t * 16;
      const __bf16* krow = &kt[buf][(t * 16 + lm) * KSTRIDE];
      v8f acc = {};
#pragma unroll
      for (int ks = 0; ks < 8; ++ks) {
        v16bf b = load_frag_bf16(krow, ks * 32, half);
        acc = wmma_bf16(qf[ks], b, acc);
      }
#pragma unroll
      for (int r = 0; r < 8; ++r) {
        float v = acc[r];
        logits[(size_t)(m0 + r + 8 * half) * 4096 + n0 + lm] = v;
        if (v > best[r]) { best[r] = v; bidx[r] = n0 + lm; }  // strict >: first max
      }
    }

    wait_async_lds();
    __syncthreads();
  }

  // reduce (val,idx) across 16 lanes per row; ties -> lower idx (jnp argmax)
#pragma unroll
  for (int r = 0; r < 8; ++r) {
    float bv = best[r];
    int   bi = bidx[r];
#pragma unroll
    for (int off = 1; off < 16; off <<= 1) {
      float ov = __shfl_xor(bv, off, 32);
      int   oi = __shfl_xor(bi, off, 32);
      if (ov > bv || (ov == bv && oi < bi)) { bv = ov; bi = oi; }
    }
    if (lm == 0) amax[m0 + r + 8 * half] = bi;
  }
}

// ---------------------------------------------------------------------------
// Kernel 5: z_q = z_q_2 = v[idx]; emit idx (as float) per token.
__global__ void k_finalize(const int* __restrict__ amax, const float* __restrict__ v,
                           float* __restrict__ out_idx, float* __restrict__ zq,
                           float* __restrict__ zq2) {
  const int t = blockIdx.x;
  const int c = threadIdx.x;
  const int idx = amax[t];
  const float val = v[(size_t)idx * 256 + c];
  zq[(size_t)t * 256 + c]  = val;
  zq2[(size_t)t * 256 + c] = val;
  if (c == 0) out_idx[t] = (float)idx;
}

// ---------------------------------------------------------------------------
extern "C" void kernel_launch(void* const* d_in, const int* in_sizes, int n_in,
                              void* d_out, int out_size, void* d_ws, size_t ws_size,
                              hipStream_t stream) {
  const float* hs = (const float*)d_in[0];   // [32768,256]
  const float* cb = (const float*)d_in[1];   // [4096,256]
  const float* wq = (const float*)d_in[2];
  const float* bq = (const float*)d_in[3];
  const float* wk = (const float*)d_in[4];
  const float* bk = (const float*)d_in[5];
  const float* wv = (const float*)d_in[6];
  const float* bv = (const float*)d_in[7];
  const float* gq = (const float*)d_in[8];
  const float* gk = (const float*)d_in[9];

  char* ws = (char*)d_ws;
  __bf16* wTq = (__bf16*)(ws);                       // 128 KB
  __bf16* wTk = (__bf16*)(ws + 131072);              // 128 KB
  __bf16* wTv = (__bf16*)(ws + 262144);              // 128 KB
  __bf16* qb  = (__bf16*)(ws + 393216);              // 16 MB
  __bf16* kbm = (__bf16*)(ws + 393216 + 16777216);   // 2 MB
  float*  vws = (float*) (ws + 393216 + 16777216 + 2097152);           // 4 MB
  int*    amx = (int*)   (ws + 393216 + 16777216 + 2097152 + 4194304); // 128 KB

  float* out     = (float*)d_out;
  float* logits  = out;                                    // [32768,4096]
  float* out_idx = out + (size_t)32768 * 4096;             // [32768,1]
  float* zq      = out_idx + 32768;                        // [32768,256]
  float* zq2     = zq + (size_t)32768 * 256;               // [32768,256]

  k_convert_w<<<768, 256, 0, stream>>>(wq, wk, wv, wTq, wTk, wTv);
  // scale 1/sqrt(256) = 1/16 folded exactly into bf16 q
  k_proj_norm<<<2048, 256, 0, stream>>>(hs, wTq, bq, gq, qb, 1.0f / 16.0f);
  k_proj_norm<<<256, 256, 0, stream>>>(cb, wTk, bk, gk, kbm, 1.0f);
  k_proj_plain<<<256, 256, 0, stream>>>(cb, wTv, bv, vws);
  k_logits<<<256, 256, 0, stream>>>(qb, kbm, logits, amx);
  k_finalize<<<32768, 256, 0, stream>>>(amx, vws, out_idx, zq, zq2);
}
